// TransformerLayer_54906861912355
// MI455X (gfx1250) — matrix operand order repaired and checked by reading, not compiled, not run on hardware
//
#include <hip/hip_runtime.h>
#include <hip/hip_bf16.h>

typedef __attribute__((ext_vector_type(16))) _Float16 v16h;
typedef __attribute__((ext_vector_type(8)))  _Float16 v8h;
typedef __attribute__((ext_vector_type(8)))  float    v8f;

#define BATCH 4
#define SEQ   4096
#define HID   256
#define FFD   1024
#define BS    (BATCH * SEQ)   // 16384 rows

// ---------------------------------------------------------------------------
// WMMA helper: D = A(16x32 f16) * B(32x16 f16) + C(16x16 f32)
// ---------------------------------------------------------------------------
__device__ __forceinline__ v8f wmma_f16(v16h a, v16h b, v8f c) {
    return __builtin_amdgcn_wmma_f32_16x16x32_f16(
        /*neg_a=*/false, a, /*neg_b=*/false, b,
        /*c_mod=*/(short)0, c, /*reuse_a=*/false, /*reuse_b=*/false);
}

// A-fragment (16x32, f16) from row-major f16 source. `base` points at tile
// origin (row0*ld + k0). Lane layout per ISA 7.12.2:
//   lane = {m(0..15), halfwave g}, halves i: k = (i>>3)*16 + g*8 + (i&7)
__device__ __forceinline__ v16h load_a16(const _Float16* __restrict__ base,
                                         int ld, int lane) {
    int m = lane & 15;
    int g = lane >> 4;
    const _Float16* p = base + (size_t)m * ld + g * 8;
    v8h lo = *reinterpret_cast<const v8h*>(p);        // k = g*8 .. g*8+7
    v8h hi = *reinterpret_cast<const v8h*>(p + 16);   // k = 16+g*8 ..
    v16h r;
#pragma unroll
    for (int i = 0; i < 8; ++i) { r[i] = lo[i]; r[i + 8] = hi[i]; }
    return r;
}

// B-fragment (32x16, f16) from row-major f16 source. `base` points at tile
// origin (k0*ld + n0). Lane holds column n=lane&15, halves i: k = (i>>3)*16 + g*8 + (i&7).
__device__ __forceinline__ v16h load_b16(const _Float16* __restrict__ base,
                                         int ld, int lane) {
    int n = lane & 15;
    int g = lane >> 4;
    const _Float16* p = base + (size_t)(g * 8) * ld + n;
    v16h r;
#pragma unroll
    for (int i = 0; i < 8; ++i) { r[i] = p[(size_t)i * ld]; r[i + 8] = p[(size_t)(16 + i) * ld]; }
    return r;
}

// ---------------------------------------------------------------------------
// f32 -> f16 conversion (prep for all WMMA operands)
// ---------------------------------------------------------------------------
__global__ void conv_f32_f16(const float* __restrict__ in,
                             _Float16* __restrict__ out, int n) {
    int i = blockIdx.x * blockDim.x + threadIdx.x;
    if (i < n) out[i] = (_Float16)in[i];
}

// ---------------------------------------------------------------------------
// QKV projection: q/k/v = x @ W + b. K is written TRANSPOSED (kT[b][h][s])
// so the attention score B-fragments are contiguous loads.
// One wave -> one 16x16 output tile. 3 * 1024 * 16 = 49152 tiles.
// ---------------------------------------------------------------------------
__global__ __launch_bounds__(256) void qkv_kernel(
    const _Float16* __restrict__ x_h,
    const _Float16* __restrict__ wq_h, const _Float16* __restrict__ wk_h,
    const _Float16* __restrict__ wv_h,
    const float* __restrict__ bq, const float* __restrict__ bk,
    const float* __restrict__ bv,
    _Float16* __restrict__ q_h, _Float16* __restrict__ kT_h,
    _Float16* __restrict__ v_h) {
    int lane  = threadIdx.x & 31;
    int gw    = blockIdx.x * 8 + (threadIdx.x >> 5);
    int which = gw / (1024 * 16);          // 0=q, 1=k, 2=v
    int rem   = gw % (1024 * 16);
    int mt    = rem >> 4;                  // row tile over B*S (0..1023)
    int nt    = rem & 15;                  // col tile over H   (0..15)

    const _Float16* w   = (which == 0) ? wq_h : (which == 1) ? wk_h : wv_h;
    const float*    bia = (which == 0) ? bq   : (which == 1) ? bk   : bv;

    v8f acc = {};
#pragma unroll
    for (int c = 0; c < 8; ++c) {          // K = 256 in chunks of 32
        v16h a = load_a16(x_h + (size_t)(mt * 16) * HID + c * 32, HID, lane);
        v16h b = load_b16(w + (size_t)(c * 32) * HID + nt * 16, HID, lane);
        acc = wmma_f16(a, b, acc);
    }

    int   coln = nt * 16 + (lane & 15);
    float bval = bia[coln];
#pragma unroll
    for (int j = 0; j < 8; ++j) {
        int   r = mt * 16 + j + ((lane >> 4) << 3);
        float f = acc[j] + bval;
        if (which == 0) {
            q_h[(size_t)r * HID + coln] = (_Float16)f;
        } else if (which == 2) {
            v_h[(size_t)r * HID + coln] = (_Float16)f;
        } else {
            int bb = r >> 12, s = r & 4095;
            kT_h[((size_t)bb * HID + coln) * SEQ + s] = (_Float16)f;
        }
    }
}

// ---------------------------------------------------------------------------
// Flash attention: one wave owns 16 query rows. Online softmax over key tiles
// of 32; accumulates 16x256 output in registers (16 C-tiles).
// Grid: BATCH * SEQ/64 blocks of 128 threads (4 waves).
// ---------------------------------------------------------------------------
__global__ __launch_bounds__(128) void attn_kernel(
    const _Float16* __restrict__ q_h, const _Float16* __restrict__ kT_h,
    const _Float16* __restrict__ v_h, float* __restrict__ attn_f,
    _Float16* __restrict__ attn_h) {
    __shared__ _Float16 pbuf[4][16 * 32];   // per-wave P-tile transpose buffer

    int lane = threadIdx.x & 31;
    int wv   = threadIdx.x >> 5;
    int b    = blockIdx.x >> 6;                       // 0..3
    int qt   = ((blockIdx.x & 63) << 2) | wv;         // 0..255
    int s0   = qt * 16;

    const _Float16* qb  = q_h  + (size_t)b * SEQ * HID;
    const _Float16* kTb = kT_h + (size_t)b * HID * SEQ;
    const _Float16* vb  = v_h  + (size_t)b * SEQ * HID;
    _Float16* myp = &pbuf[wv][0];

    // Q row-block fragments, K = 0..255 (held for the whole pass)
    v16h qf[8];
#pragma unroll
    for (int c = 0; c < 8; ++c)
        qf[c] = load_a16(qb + (size_t)s0 * HID + c * 32, HID, lane);

    v8f acc[16];
#pragma unroll
    for (int t = 0; t < 16; ++t) acc[t] = {};
    float mrow[8], lrow[8];
#pragma unroll
    for (int j = 0; j < 8; ++j) { mrow[j] = -1e30f; lrow[j] = 0.0f; }

#pragma unroll 1
    for (int kt = 0; kt < SEQ; kt += 32) {
        // ---- scores: 16 queries x 32 keys, contraction over H=256 ----
        v8f sa = {}, sb = {};
#pragma unroll
        for (int c = 0; c < 8; ++c) {
            v16h kf0 = load_b16(kTb + (size_t)(c * 32) * SEQ + kt, SEQ, lane);
            sa = wmma_f16(qf[c], kf0, sa);
            v16h kf1 = load_b16(kTb + (size_t)(c * 32) * SEQ + kt + 16, SEQ, lane);
            sb = wmma_f16(qf[c], kf1, sb);
        }
        // ---- online softmax (C-layout: lane=N col, vgpr j = row) ----
#pragma unroll
        for (int j = 0; j < 8; ++j) {
            float a0 = sa[j] * 0.0625f;   // 1/sqrt(256)
            float a1 = sb[j] * 0.0625f;
            float mx = fmaxf(a0, a1);
            mx = fmaxf(mx, __shfl_xor(mx, 1, 16));
            mx = fmaxf(mx, __shfl_xor(mx, 2, 16));
            mx = fmaxf(mx, __shfl_xor(mx, 4, 16));
            mx = fmaxf(mx, __shfl_xor(mx, 8, 16));
            float mnew  = fmaxf(mrow[j], mx);
            float alpha = __expf(mrow[j] - mnew);
            mrow[j] = mnew;
            float p0 = __expf(a0 - mnew);
            float p1 = __expf(a1 - mnew);
            float ps = p0 + p1;
            ps += __shfl_xor(ps, 1, 16);
            ps += __shfl_xor(ps, 2, 16);
            ps += __shfl_xor(ps, 4, 16);
            ps += __shfl_xor(ps, 8, 16);
            lrow[j] = lrow[j] * alpha + ps;
#pragma unroll
            for (int t = 0; t < 16; ++t) acc[t][j] *= alpha;
            // stash probabilities (f16) for layout transpose via LDS
            int r = j + ((lane >> 4) << 3);
            int c = lane & 15;
            myp[r * 32 + c]      = (_Float16)p0;
            myp[r * 32 + 16 + c] = (_Float16)p1;
        }
        // in-wave LDS fence: DS ops are in-order per wave; block reordering
        asm volatile("s_wait_dscnt 0" ::: "memory");
        __builtin_amdgcn_wave_barrier();

        // reload P as an A-fragment (16x32)
        v16h pf;
        {
            int m = lane & 15, g = lane >> 4;
            v8h lo = *reinterpret_cast<const v8h*>(&myp[m * 32 + g * 8]);
            v8h hi = *reinterpret_cast<const v8h*>(&myp[m * 32 + 16 + g * 8]);
#pragma unroll
            for (int i = 0; i < 8; ++i) { pf[i] = lo[i]; pf[i + 8] = hi[i]; }
        }
        __builtin_amdgcn_wave_barrier();

        // ---- accumulate P(16x32) * V(32x256) ----
#pragma unroll
        for (int t = 0; t < 16; ++t) {
            v16h vf = load_b16(vb + (size_t)kt * HID + t * 16, HID, lane);
            acc[t] = wmma_f16(pf, vf, acc[t]);
        }
    }

    // normalize and write out (f32 for residual, f16 for FFN GEMM input)
    float*    ab  = attn_f + (size_t)b * SEQ * HID;
    _Float16* ahb = attn_h + (size_t)b * SEQ * HID;
#pragma unroll
    for (int j = 0; j < 8; ++j) {
        float inv = 1.0f / lrow[j];
        int   r   = s0 + j + ((lane >> 4) << 3);
#pragma unroll
        for (int t = 0; t < 16; ++t) {
            int   col = t * 16 + (lane & 15);
            float o   = acc[t][j] * inv;
            ab[(size_t)r * HID + col]  = o;
            ahb[(size_t)r * HID + col] = (_Float16)o;
        }
    }
}

// ---------------------------------------------------------------------------
// FFN1: h = relu(attn @ w1 + b1), stored f16. 1024 x 64 tiles.
// ---------------------------------------------------------------------------
__global__ __launch_bounds__(256) void ffn1_kernel(
    const _Float16* __restrict__ a_h, const _Float16* __restrict__ w1_h,
    const float* __restrict__ b1, _Float16* __restrict__ h_h) {
    int lane = threadIdx.x & 31;
    int gw   = blockIdx.x * 8 + (threadIdx.x >> 5);
    int mt   = gw >> 6;     // 0..1023
    int nt   = gw & 63;     // 0..63

    v8f acc = {};
#pragma unroll
    for (int c = 0; c < 8; ++c) {
        v16h a = load_a16(a_h + (size_t)(mt * 16) * HID + c * 32, HID, lane);
        v16h b = load_b16(w1_h + (size_t)(c * 32) * FFD + nt * 16, FFD, lane);
        acc = wmma_f16(a, b, acc);
    }
    int   col  = nt * 16 + (lane & 15);
    float bval = b1[col];
#pragma unroll
    for (int j = 0; j < 8; ++j) {
        int   r = mt * 16 + j + ((lane >> 4) << 3);
        float f = acc[j] + bval;
        f = (f > 0.0f) ? f : 0.0f;
        h_h[(size_t)r * FFD + col] = (_Float16)f;
    }
}

// ---------------------------------------------------------------------------
// FFN2 + residual: y = attn + h @ w2 + b2 (f32). 1024 x 16 tiles.
// ---------------------------------------------------------------------------
__global__ __launch_bounds__(256) void ffn2_kernel(
    const _Float16* __restrict__ h_h, const _Float16* __restrict__ w2_h,
    const float* __restrict__ b2, const float* __restrict__ attn_f,
    float* __restrict__ y) {
    int lane = threadIdx.x & 31;
    int gw   = blockIdx.x * 8 + (threadIdx.x >> 5);
    int mt   = gw >> 4;     // 0..1023
    int nt   = gw & 15;     // 0..15

    v8f acc = {};
#pragma unroll
    for (int c = 0; c < 32; ++c) {          // K = 1024
        v16h a = load_a16(h_h + (size_t)(mt * 16) * FFD + c * 32, FFD, lane);
        v16h b = load_b16(w2_h + (size_t)(c * 32) * HID + nt * 16, HID, lane);
        acc = wmma_f16(a, b, acc);
    }
    int   col  = nt * 16 + (lane & 15);
    float bval = b2[col];
#pragma unroll
    for (int j = 0; j < 8; ++j) {
        int r = mt * 16 + j + ((lane >> 4) << 3);
        y[(size_t)r * HID + col] =
            attn_f[(size_t)r * HID + col] + acc[j] + bval;
    }
}

// ---------------------------------------------------------------------------
// LayerNorm over H=256: one block (256 threads) per row.
// ---------------------------------------------------------------------------
__global__ __launch_bounds__(256) void ln_kernel(
    const float* __restrict__ y, const float* __restrict__ g,
    const float* __restrict__ be, float* __restrict__ out) {
    __shared__ float red[16];
    int   row = blockIdx.x;
    int   t   = threadIdx.x;
    float v   = y[(size_t)row * HID + t];
    float s = v, s2 = v * v;
#pragma unroll
    for (int off = 16; off >= 1; off >>= 1) {
        s  += __shfl_xor(s, off, 32);
        s2 += __shfl_xor(s2, off, 32);
    }
    if ((t & 31) == 0) { red[t >> 5] = s; red[8 + (t >> 5)] = s2; }
    __syncthreads();
    float ts = 0.0f, ts2 = 0.0f;
#pragma unroll
    for (int i = 0; i < 8; ++i) { ts += red[i]; ts2 += red[8 + i]; }
    float mu  = ts * (1.0f / HID);
    float var = ts2 * (1.0f / HID) - mu * mu;
    out[(size_t)row * HID + t] = (v - mu) * rsqrtf(var + 1e-5f) * g[t] + be[t];
}

// ---------------------------------------------------------------------------
extern "C" void kernel_launch(void* const* d_in, const int* in_sizes, int n_in,
                              void* d_out, int out_size, void* d_ws,
                              size_t ws_size, hipStream_t stream) {
    const float* x    = (const float*)d_in[0];
    const float* wq   = (const float*)d_in[1];
    const float* bq   = (const float*)d_in[2];
    const float* wk   = (const float*)d_in[3];
    const float* bk   = (const float*)d_in[4];
    const float* wvp  = (const float*)d_in[5];
    const float* bv   = (const float*)d_in[6];
    const float* w1   = (const float*)d_in[7];
    const float* b1   = (const float*)d_in[8];
    const float* w2   = (const float*)d_in[9];
    const float* b2   = (const float*)d_in[10];
    const float* ln_g = (const float*)d_in[11];
    const float* ln_b = (const float*)d_in[12];
    float*       out  = (float*)d_out;

    // ---- workspace layout (all 16B aligned) ----
    char* wsp = (char*)d_ws;
    _Float16* x_h   = (_Float16*)wsp; wsp += (size_t)BS * HID * 2;   // 8 MB
    _Float16* q_h   = (_Float16*)wsp; wsp += (size_t)BS * HID * 2;
    _Float16* kT_h  = (_Float16*)wsp; wsp += (size_t)BS * HID * 2;   // [b][h][s]
    _Float16* v_h   = (_Float16*)wsp; wsp += (size_t)BS * HID * 2;
    _Float16* a_h   = (_Float16*)wsp; wsp += (size_t)BS * HID * 2;
    _Float16* h_h   = (_Float16*)wsp; wsp += (size_t)BS * FFD * 2;   // 32 MB
    _Float16* wq_h  = (_Float16*)wsp; wsp += (size_t)HID * HID * 2;
    _Float16* wk_h  = (_Float16*)wsp; wsp += (size_t)HID * HID * 2;
    _Float16* wv_h  = (_Float16*)wsp; wsp += (size_t)HID * HID * 2;
    _Float16* w1_h  = (_Float16*)wsp; wsp += (size_t)HID * FFD * 2;
    _Float16* w2_h  = (_Float16*)wsp; wsp += (size_t)FFD * HID * 2;
    float*    at_f  = (float*)wsp;    wsp += (size_t)BS * HID * 4;   // 16 MB
    float*    y_f   = (float*)wsp;    wsp += (size_t)BS * HID * 4;

    // 1. convert operands to f16
    conv_f32_f16<<<(BS * HID) / 256, 256, 0, stream>>>(x, x_h, BS * HID);
    conv_f32_f16<<<(HID * HID) / 256, 256, 0, stream>>>(wq, wq_h, HID * HID);
    conv_f32_f16<<<(HID * HID) / 256, 256, 0, stream>>>(wk, wk_h, HID * HID);
    conv_f32_f16<<<(HID * HID) / 256, 256, 0, stream>>>(wvp, wv_h, HID * HID);
    conv_f32_f16<<<(HID * FFD) / 256, 256, 0, stream>>>(w1, w1_h, HID * FFD);
    conv_f32_f16<<<(FFD * HID) / 256, 256, 0, stream>>>(w2, w2_h, FFD * HID);

    // 2. QKV projections: 3 * 1024 * 16 tiles, 8 waves/block
    qkv_kernel<<<(3 * 1024 * 16) / 8, 256, 0, stream>>>(
        x_h, wq_h, wk_h, wv_h, bq, bk, bv, q_h, kT_h, v_h);

    // 3. flash attention: BATCH * SEQ/64 blocks of 4 waves
    attn_kernel<<<BATCH * (SEQ / 64), 128, 0, stream>>>(q_h, kT_h, v_h, at_f,
                                                        a_h);

    // 4. FFN
    ffn1_kernel<<<(1024 * 64) / 8, 256, 0, stream>>>(a_h, w1_h, b1, h_h);
    ffn2_kernel<<<(1024 * 16) / 8, 256, 0, stream>>>(h_h, w2_h, b2, at_f, y_f);

    // 5. layernorm -> d_out
    ln_kernel<<<BS, 256, 0, stream>>>(y_f, ln_g, ln_b, out);
}
